// ViT_10522669875675
// MI455X (gfx1250) — matrix operand order repaired and checked by reading, not compiled, not run on hardware
//
#include <hip/hip_runtime.h>
#include <hip/hip_bf16.h>
#include <cstdint>

// ---------------------------------------------------------------------------
// ViT-Base forward for MI455X (gfx1250, wave32, WMMA).
// f16 operands + f32 accumulation via V_WMMA_F32_16X16X32_F16.
// A-tile staging uses GLOBAL_LOAD_ASYNC_TO_LDS_B128 (ASYNCcnt) via inline asm.
// ---------------------------------------------------------------------------

typedef __attribute__((ext_vector_type(16))) _Float16 v16h;
typedef __attribute__((ext_vector_type(8)))  _Float16 v8h;
typedef __attribute__((ext_vector_type(8)))  float    v8f;

#define D_     768
#define H_     12
#define L_     12
#define F_     3072
#define S_     197
#define B_     32
#define NP_    196
#define DH_    64
#define ROWS_  (B_*S_)    // 6304
#define PROWS_ (B_*NP_)   // 6272
#define NCLS_  1000

__device__ __forceinline__ v16h mk16(v8h lo, v8h hi) {
  return __builtin_shufflevector(lo, hi, 0,1,2,3,4,5,6,7,8,9,10,11,12,13,14,15);
}
__device__ __forceinline__ v8f wmma_f16(v16h a, v16h b, v8f c) {
  // (neg_a, A, neg_b, B, c_mod, C, reuse_a, reuse_b)
  return __builtin_amdgcn_wmma_f32_16x16x32_f16(false, a, false, b, (short)0, c, false, false);
}
__device__ __forceinline__ float gelu_f(float x) {
  return 0.5f * x * (1.f + erff(x * 0.70710678118654752f));
}
// flat shared pointer -> LDS byte offset (aperture rule: LDS offset is the low 32 bits of the address)
__device__ __forceinline__ uint32_t lds_off(const void* p) {
  return (uint32_t)(uintptr_t)p;
}
// per-lane async global->LDS copy of 16B (+16B at offset 16), ASYNCcnt-tracked
__device__ __forceinline__ void async_copy32B(uint32_t lds_byte, const void* gaddr) {
  asm volatile("global_load_async_to_lds_b128 %0, %1, off"
               :: "v"(lds_byte), "v"(gaddr) : "memory");
  asm volatile("global_load_async_to_lds_b128 %0, %1, off offset:16"
               :: "v"(lds_byte), "v"(gaddr) : "memory");
}
__device__ __forceinline__ void wait_async0() {
  asm volatile("s_wait_asynccnt 0x0" ::: "memory");
}

// ---------------------------------------------------------------------------
// Elementwise helpers
// ---------------------------------------------------------------------------
__global__ void k_cvt(const float* __restrict__ src, _Float16* __restrict__ dst, size_t n) {
  size_t i = (size_t)blockIdx.x * 256 + threadIdx.x;
  if (i < n) dst[i] = (_Float16)src[i];
}

// (L,H,D,DH) f32  ->  (L, D, H*DH) f16   so GEMM column n = h*64+e
__global__ void k_pack_qkv(const float* __restrict__ src, _Float16* __restrict__ dst) {
  size_t i = (size_t)blockIdx.x * 256 + threadIdx.x;
  const size_t n = (size_t)L_ * D_ * D_;
  if (i >= n) return;
  int l  = (int)(i / (size_t)(D_ * D_));
  int r  = (int)(i % (size_t)(D_ * D_));
  int d  = r / D_;
  int nn = r % D_;
  int h  = nn / DH_;
  int e  = nn % DH_;
  dst[i] = (_Float16)src[(((size_t)l * H_ + h) * D_ + d) * DH_ + e];
}

// x (B,C,224,224) f32 -> patches (B*196, 768) f16
__global__ void k_patchify(const float* __restrict__ x, _Float16* __restrict__ pa) {
  size_t i = (size_t)blockIdx.x * 256 + threadIdx.x;
  const size_t n = (size_t)PROWS_ * D_;
  if (i >= n) return;
  int k   = (int)(i % D_);          // c*256 + py*16 + px
  size_t row = i / D_;              // b*196 + np
  int np  = (int)(row % NP_);
  int b   = (int)(row / NP_);
  int c   = k >> 8;
  int py  = (k >> 4) & 15;
  int px  = k & 15;
  int gy  = np / 14, gx = np % 14;
  pa[i] = (_Float16)x[(((size_t)b * 3 + c) * 224 + gy * 16 + py) * 224 + gx * 16 + px];
}

// h[b,0,:] = cls + pos[0]; h[b,1+np,:] = pe[b,np] + pos[1+np]
__global__ void k_assemble(const float* __restrict__ pe, const float* __restrict__ cls,
                           const float* __restrict__ pos, float* __restrict__ h) {
  size_t i = (size_t)blockIdx.x * 256 + threadIdx.x;
  const size_t n = (size_t)ROWS_ * D_;
  if (i >= n) return;
  int c = (int)(i % D_);
  size_t row = i / D_;
  int s = (int)(row % S_);
  int b = (int)(row / S_);
  float v = (s == 0) ? cls[c] : pe[((size_t)b * NP_ + (s - 1)) * D_ + c];
  h[i] = v + pos[(size_t)s * D_ + c];
}

// ---------------------------------------------------------------------------
// LayerNorm over last dim (768), f32 in (strided rows) -> f16 out (dense rows)
// ---------------------------------------------------------------------------
__global__ __launch_bounds__(256) void k_layernorm(
    const float* __restrict__ x, size_t row_stride,
    const float* __restrict__ g, const float* __restrict__ bta,
    _Float16* __restrict__ out) {
  __shared__ float red[8];
  const int row = blockIdx.x;
  const int tid = threadIdx.x, lane = tid & 31, wv = tid >> 5;
  const float* xr = x + (size_t)row * row_stride;
  float v0 = xr[tid], v1 = xr[tid + 256], v2 = xr[tid + 512];
  float s = v0 + v1 + v2;
  #pragma unroll
  for (int d = 16; d; d >>= 1) s += __shfl_xor(s, d, 32);
  if (lane == 0) red[wv] = s;
  __syncthreads();
  float tot = 0.f;
  #pragma unroll
  for (int i = 0; i < 8; i++) tot += red[i];
  const float mean = tot * (1.f / 768.f);
  __syncthreads();
  float d0 = v0 - mean, d1 = v1 - mean, d2 = v2 - mean;
  float q = d0 * d0 + d1 * d1 + d2 * d2;
  #pragma unroll
  for (int d = 16; d; d >>= 1) q += __shfl_xor(q, d, 32);
  if (lane == 0) red[wv] = q;
  __syncthreads();
  float vt = 0.f;
  #pragma unroll
  for (int i = 0; i < 8; i++) vt += red[i];
  const float rstd = rsqrtf(vt * (1.f / 768.f) + 1e-5f);
  _Float16* orow = out + (size_t)row * D_;
  orow[tid]       = (_Float16)(d0 * rstd * g[tid]       + bta[tid]);
  orow[tid + 256] = (_Float16)(d1 * rstd * g[tid + 256] + bta[tid + 256]);
  orow[tid + 512] = (_Float16)(d2 * rstd * g[tid + 512] + bta[tid + 512]);
}

// ---------------------------------------------------------------------------
// f16 WMMA GEMM: C(MxN) = A(MxK) @ B(KxN)  (+bias [+gelu|+resid])
// Block tile 128x128, BK=32; 8 waves, each owns 32x64 = 2x4 WMMA frags
// (8 v_wmma per K step per wave). A tile staged with async-to-LDS.
// EPI: 0 = bias, 1 = bias+gelu, 2 = bias+resid.
// ---------------------------------------------------------------------------
template<int EPI, bool SF32, bool SF16>
__global__ __launch_bounds__(256) void k_gemm(
    const _Float16* __restrict__ A, const _Float16* __restrict__ B,
    const float* __restrict__ bias, const float* __restrict__ resid,
    float* __restrict__ Cf, _Float16* __restrict__ Ch,
    int M, int N, int K) {
  __shared__ _Float16 As[128][40];   // row-major A tile (pad 32 -> 40)
  __shared__ _Float16 Bt[128][40];   // transposed B tile: [n][k]

  const int tid  = threadIdx.x;
  const int lane = tid & 31;
  const int wv   = tid >> 5;
  const int wm   = wv >> 1;          // 0..3 : 32-row group
  const int wn   = wv & 1;           // 0..1 : 64-col group
  const int m0   = blockIdx.y * 128;
  const int n0   = blockIdx.x * 128;
  const int lrow = lane & 15;
  const int la   = (lane < 16) ? 0 : 8;    // A-frag K split (0-7/16-23 vs 8-15/24-31)
  const int lb   = (lane < 16) ? 0 : 16;   // unused (B frags use the la K split, same as A)

  // A staging: thread -> one 32B chunk (row, 16-half column group)
  const int sa_r = tid >> 1, sa_c = (tid & 1) * 16;
  const int sa_g = m0 + sa_r;
  const bool sa_ok = sa_g < M;
  const _Float16* Ab = A + (size_t)sa_g * K + sa_c;
  const uint32_t As_lds = lds_off(&As[sa_r][sa_c]);
  // B staging: thread -> one k-row, 16 consecutive n
  const int sb_k = tid >> 3, sb_n = (tid & 7) * 16;
  const _Float16* Bb = B + (size_t)sb_k * N + n0 + sb_n;

  v8f acc[2][4];
  #pragma unroll
  for (int i = 0; i < 2; i++)
    #pragma unroll
    for (int j = 0; j < 4; j++)
      #pragma unroll
      for (int e = 0; e < 8; e++) acc[i][j][e] = 0.f;

  for (int k0 = 0; k0 < K; k0 += 32) {
    // ---- stage A (128x32 halves) via async global->LDS ----
    if (sa_ok) {
      async_copy32B(As_lds, Ab + k0);
    } else {
      #pragma unroll
      for (int i = 0; i < 16; i++) As[sa_r][sa_c + i] = (_Float16)0.f;
    }
    // ---- stage B transposed (32x128 -> [n][k]) ----
    {
      const _Float16* src = Bb + (size_t)k0 * N;
      if (n0 + 128 <= N) {
        v8h b0 = *(const v8h*)src;
        v8h b1 = *(const v8h*)(src + 8);
        #pragma unroll
        for (int j = 0; j < 8; j++) Bt[sb_n + j][sb_k]     = b0[j];
        #pragma unroll
        for (int j = 0; j < 8; j++) Bt[sb_n + 8 + j][sb_k] = b1[j];
      } else {
        #pragma unroll
        for (int j = 0; j < 16; j++) {
          int n = n0 + sb_n + j;
          Bt[sb_n + j][sb_k] = (n < N) ? src[j] : (_Float16)0.f;
        }
      }
    }
    if (k0 + 32 < K)   // CDNA5 global_prefetch of next B K-tile
      __builtin_prefetch(Bb + (size_t)(k0 + 32) * N, 0, 1);
    wait_async0();
    __syncthreads();

    v16h af[2], bf[4];
    #pragma unroll
    for (int t = 0; t < 2; t++) {
      const _Float16* p = &As[wm * 32 + t * 16 + lrow][0];
      af[t] = mk16(*(const v8h*)(p + la), *(const v8h*)(p + la + 16));
    }
    #pragma unroll
    for (int t = 0; t < 4; t++) {
      const _Float16* p = &Bt[wn * 64 + t * 16 + lrow][0];
      bf[t] = mk16(*(const v8h*)(p + la), *(const v8h*)(p + la + 16));
    }
    #pragma unroll
    for (int i = 0; i < 2; i++)
      #pragma unroll
      for (int j = 0; j < 4; j++)
        acc[i][j] = wmma_f16(af[i], bf[j], acc[i][j]);
    __syncthreads();
  }

  // epilogue: C/D layout -> VGPR e: M = e (lanes 0-15) / 8+e (lanes 16-31), N = lane&15
  #pragma unroll
  for (int i = 0; i < 2; i++) {
    const int rb = m0 + wm * 32 + i * 16 + ((lane < 16) ? 0 : 8);
    #pragma unroll
    for (int j = 0; j < 4; j++) {
      const int col = n0 + wn * 64 + j * 16 + lrow;
      if (col >= N) continue;
      const float bvv = bias ? bias[col] : 0.f;
      #pragma unroll
      for (int e = 0; e < 8; e++) {
        const int r = rb + e;
        if (r >= M) continue;
        float v = acc[i][j][e] + bvv;
        if (EPI == 1) v = gelu_f(v);
        if (EPI == 2) v += resid[(size_t)r * N + col];
        if (SF32) Cf[(size_t)r * N + col] = v;
        if (SF16) Ch[(size_t)r * N + col] = (_Float16)v;
      }
    }
  }
}

// ---------------------------------------------------------------------------
// Attention: one block per (b, h, 32-query-row tile). Q/K/V stored as
// (b*S, h*64+e) f16 with row stride 768. WMMA for QK^T and P@V.
// ---------------------------------------------------------------------------
__global__ __launch_bounds__(256) void k_attention(
    const _Float16* __restrict__ Q, const _Float16* __restrict__ Km,
    const _Float16* __restrict__ V, _Float16* __restrict__ O) {
  __shared__ _Float16 Qs[32][72];    // 32 x 64 Q tile
  __shared__ float    Ss[32][228];   // scores, cols 0..223
  __shared__ _Float16 Ps[32][240];   // softmax probs, zero-padded to 224
  __shared__ _Float16 Vt[64][40];    // transposed V tile [e][t]

  const int bh = blockIdx.y;
  const int b  = bh / H_;
  const int hh = bh % H_;
  const int m0 = blockIdx.x * 32;
  const int tid = threadIdx.x, lane = tid & 31, wv = tid >> 5;
  const int wm = wv >> 2;            // 0..1
  const int wn = wv & 3;             // 0..3
  const int lrow = lane & 15;
  const int la = (lane < 16) ? 0 : 8;
  const int lb = (lane < 16) ? 0 : 16;
  const size_t base = (size_t)b * S_ * D_ + (size_t)hh * DH_;

  // stage Q (32 x 64 halves)
  {
    const int r = tid >> 3, c = (tid & 7) * 8;
    const int gr = m0 + r;
    if (gr < S_) {
      *(v8h*)&Qs[r][c] = *(const v8h*)(Q + base + (size_t)gr * D_ + c);
    } else {
      #pragma unroll
      for (int i = 0; i < 8; i++) Qs[r][c + i] = (_Float16)0.f;
    }
  }
  __syncthreads();

  // phase 1: scores = (Q K^T) * 0.125, masked past S
  for (int kt = 0; kt < 4; ++kt) {
    const int t0 = kt * 64 + wn * 16;
    const int tg = t0 + lrow;
    v16h bf0, bf1;
    if (tg < S_) {
      const _Float16* kp = Km + base + (size_t)tg * D_;
      bf0 = mk16(*(const v8h*)(kp + la),      *(const v8h*)(kp + la + 16));
      bf1 = mk16(*(const v8h*)(kp + 32 + la), *(const v8h*)(kp + 32 + la + 16));
    } else {
      #pragma unroll
      for (int i = 0; i < 16; i++) { bf0[i] = (_Float16)0.f; bf1[i] = (_Float16)0.f; }
    }
    const _Float16* qp = &Qs[wm * 16 + lrow][0];
    v16h af0 = mk16(*(const v8h*)(qp + la),      *(const v8h*)(qp + la + 16));
    v16h af1 = mk16(*(const v8h*)(qp + 32 + la), *(const v8h*)(qp + 32 + la + 16));
    v8f acc;
    #pragma unroll
    for (int e = 0; e < 8; e++) acc[e] = 0.f;
    acc = wmma_f16(af0, bf0, acc);
    acc = wmma_f16(af1, bf1, acc);
    const int rb = wm * 16 + ((lane < 16) ? 0 : 8);
    const int cc = t0 + lrow;
    if (cc < 224) {
      #pragma unroll
      for (int e = 0; e < 8; e++)
        Ss[rb + e][cc] = (cc < S_) ? acc[e] * 0.125f : -1e30f;
    }
  }
  __syncthreads();

  // phase 2: softmax, wave wv owns rows wv*4..wv*4+3
  for (int rr = 0; rr < 4; ++rr) {
    const int r = wv * 4 + rr;
    const int grow = m0 + r;
    float mx = -1e30f;
    for (int j = lane; j < S_; j += 32) mx = fmaxf(mx, Ss[r][j]);
    #pragma unroll
    for (int d = 16; d; d >>= 1) mx = fmaxf(mx, __shfl_xor(mx, d, 32));
    float sum = 0.f;
    for (int j = lane; j < S_; j += 32) {
      float e = __expf(Ss[r][j] - mx);
      Ss[r][j] = e;
      sum += e;
    }
    #pragma unroll
    for (int d = 16; d; d >>= 1) sum += __shfl_xor(sum, d, 32);
    const float inv = (grow < S_) ? 1.f / sum : 0.f;
    for (int j = lane; j < 224; j += 32)
      Ps[r][j] = (_Float16)((j < S_) ? Ss[r][j] * inv : 0.f);
  }
  __syncthreads();

  // phase 3: O = P(32x224) @ V(224x64), zero-padded K
  v8f oacc;
  #pragma unroll
  for (int e = 0; e < 8; e++) oacc[e] = 0.f;
  for (int ks = 0; ks < 7; ++ks) {
    {
      const int tl = tid >> 3, eg = (tid & 7) * 8;
      const int t = ks * 32 + tl;
      if (t < S_) {
        const _Float16* vp = V + base + (size_t)t * D_ + eg;
        #pragma unroll
        for (int j = 0; j < 8; j++) Vt[eg + j][tl] = vp[j];
      } else {
        #pragma unroll
        for (int j = 0; j < 8; j++) Vt[eg + j][tl] = (_Float16)0.f;
      }
    }
    __syncthreads();
    const _Float16* pp = &Ps[wm * 16 + lrow][ks * 32];
    v16h af = mk16(*(const v8h*)(pp + la), *(const v8h*)(pp + la + 16));
    const _Float16* vp2 = &Vt[wn * 16 + lrow][0];
    v16h bf = mk16(*(const v8h*)(vp2 + la), *(const v8h*)(vp2 + la + 16));
    oacc = wmma_f16(af, bf, oacc);
    __syncthreads();
  }
  const int rb = m0 + wm * 16 + ((lane < 16) ? 0 : 8);
  const int e0 = hh * DH_ + wn * 16 + lrow;
  #pragma unroll
  for (int e = 0; e < 8; e++) {
    const int r = rb + e;
    if (r < S_)
      O[((size_t)b * S_ + r) * D_ + e0] = (_Float16)oacc[e];
  }
}

// ---------------------------------------------------------------------------
// Host-side orchestration
// ---------------------------------------------------------------------------
extern "C" void kernel_launch(void* const* d_in, const int* in_sizes, int n_in,
                              void* d_out, int out_size, void* d_ws, size_t ws_size,
                              hipStream_t stream) {
  (void)in_sizes; (void)n_in; (void)out_size; (void)ws_size;

  const float* x    = (const float*)d_in[0];
  const float* Wp   = (const float*)d_in[1];
  const float* bp   = (const float*)d_in[2];
  const float* cls  = (const float*)d_in[3];
  const float* pos  = (const float*)d_in[4];
  const float* ln1g = (const float*)d_in[5];
  const float* ln1b = (const float*)d_in[6];
  const float* Wq   = (const float*)d_in[7];
  const float* bq   = (const float*)d_in[8];
  const float* Wk   = (const float*)d_in[9];
  const float* bk   = (const float*)d_in[10];
  const float* Wv   = (const float*)d_in[11];
  const float* bv   = (const float*)d_in[12];
  const float* Wo   = (const float*)d_in[13];
  const float* bo   = (const float*)d_in[14];
  const float* ln2g = (const float*)d_in[15];
  const float* ln2b = (const float*)d_in[16];
  const float* W1   = (const float*)d_in[17];
  const float* b1   = (const float*)d_in[18];
  const float* W2   = (const float*)d_in[19];
  const float* b2   = (const float*)d_in[20];
  const float* hlng = (const float*)d_in[21];
  const float* hlnb = (const float*)d_in[22];
  const float* hW1  = (const float*)d_in[23];
  const float* hb1  = (const float*)d_in[24];
  const float* hW2  = (const float*)d_in[25];
  const float* hb2  = (const float*)d_in[26];

  const size_t DD = (size_t)D_ * D_;        // 589824
  const size_t DF = (size_t)D_ * F_;        // 2359296

  char* wsb = (char*)d_ws;
  size_t off = 0;
  auto take = [&](size_t bytes) -> char* {
    char* p = wsb + off;
    off += (bytes + 255) & ~(size_t)255;
    return p;
  };

  // f16 weight copies
  _Float16* Wp_h  = (_Float16*)take(DD * 2);
  _Float16* Wq_h  = (_Float16*)take(L_ * DD * 2);
  _Float16* Wk_h  = (_Float16*)take(L_ * DD * 2);
  _Float16* Wv_h  = (_Float16*)take(L_ * DD * 2);
  _Float16* Wo_h  = (_Float16*)take(L_ * DD * 2);
  _Float16* W1_h  = (_Float16*)take(L_ * DF * 2);
  _Float16* W2_h  = (_Float16*)take(L_ * DF * 2);
  _Float16* hW1_h = (_Float16*)take(DD * 2);
  _Float16* hW2_h = (_Float16*)take((size_t)D_ * NCLS_ * 2);
  // activations
  float*    h_f   = (float*)   take((size_t)ROWS_ * D_ * 4);
  _Float16* xn_h  = (_Float16*)take((size_t)ROWS_ * D_ * 2);
  _Float16* q_h   = (_Float16*)take((size_t)ROWS_ * D_ * 2);
  _Float16* k_h   = (_Float16*)take((size_t)ROWS_ * D_ * 2);
  _Float16* v_h   = (_Float16*)take((size_t)ROWS_ * D_ * 2);
  _Float16* o_h   = (_Float16*)take((size_t)ROWS_ * D_ * 2);
  _Float16* mlp_h = (_Float16*)take((size_t)ROWS_ * F_ * 2);
  _Float16* pa_h  = (_Float16*)take((size_t)PROWS_ * D_ * 2);
  float*    pe_f  = (float*)   take((size_t)PROWS_ * D_ * 4);
  _Float16* cls_h = (_Float16*)take((size_t)B_ * D_ * 2);
  _Float16* c1_h  = (_Float16*)take((size_t)B_ * D_ * 2);

  const dim3 blk(256);
  auto eb = [](size_t n) { return dim3((unsigned)((n + 255) / 256)); };

  // ---- weight conversion / repack ----
  k_cvt<<<eb(DD), blk, 0, stream>>>(Wp, Wp_h, DD);
  k_pack_qkv<<<eb(L_ * DD), blk, 0, stream>>>(Wq, Wq_h);
  k_pack_qkv<<<eb(L_ * DD), blk, 0, stream>>>(Wk, Wk_h);
  k_pack_qkv<<<eb(L_ * DD), blk, 0, stream>>>(Wv, Wv_h);
  k_cvt<<<eb(L_ * DD), blk, 0, stream>>>(Wo, Wo_h, L_ * DD);
  k_cvt<<<eb(L_ * DF), blk, 0, stream>>>(W1, W1_h, L_ * DF);
  k_cvt<<<eb(L_ * DF), blk, 0, stream>>>(W2, W2_h, L_ * DF);
  k_cvt<<<eb(DD), blk, 0, stream>>>(hW1, hW1_h, DD);
  k_cvt<<<eb((size_t)D_ * NCLS_), blk, 0, stream>>>(hW2, hW2_h, (size_t)D_ * NCLS_);

  // ---- patch embedding ----
  k_patchify<<<eb((size_t)PROWS_ * D_), blk, 0, stream>>>(x, pa_h);
  k_gemm<0, true, false><<<dim3(D_ / 128, PROWS_ / 128), blk, 0, stream>>>(
      pa_h, Wp_h, bp, nullptr, pe_f, nullptr, PROWS_, D_, D_);
  k_assemble<<<eb((size_t)ROWS_ * D_), blk, 0, stream>>>(pe_f, cls, pos, h_f);

  // ---- transformer layers ----
  const dim3 gD(D_ / 128, (ROWS_ + 127) / 128);     // 6 x 50
  const dim3 gF(F_ / 128, (ROWS_ + 127) / 128);     // 24 x 50
  const dim3 gAttn((S_ + 31) / 32, B_ * H_);        // 7 x 384
  for (int l = 0; l < L_; ++l) {
    k_layernorm<<<ROWS_, blk, 0, stream>>>(h_f, (size_t)D_,
        ln1g + (size_t)l * D_, ln1b + (size_t)l * D_, xn_h);
    k_gemm<0, false, true><<<gD, blk, 0, stream>>>(
        xn_h, Wq_h + l * DD, bq + (size_t)l * D_, nullptr, nullptr, q_h, ROWS_, D_, D_);
    k_gemm<0, false, true><<<gD, blk, 0, stream>>>(
        xn_h, Wk_h + l * DD, bk + (size_t)l * D_, nullptr, nullptr, k_h, ROWS_, D_, D_);
    k_gemm<0, false, true><<<gD, blk, 0, stream>>>(
        xn_h, Wv_h + l * DD, bv + (size_t)l * D_, nullptr, nullptr, v_h, ROWS_, D_, D_);
    k_attention<<<gAttn, blk, 0, stream>>>(q_h, k_h, v_h, o_h);
    k_gemm<2, true, false><<<gD, blk, 0, stream>>>(
        o_h, Wo_h + l * DD, bo + (size_t)l * D_, h_f, h_f, nullptr, ROWS_, D_, D_);
    k_layernorm<<<ROWS_, blk, 0, stream>>>(h_f, (size_t)D_,
        ln2g + (size_t)l * D_, ln2b + (size_t)l * D_, xn_h);
    k_gemm<1, false, true><<<gF, blk, 0, stream>>>(
        xn_h, W1_h + l * DF, b1 + (size_t)l * F_, nullptr, nullptr, mlp_h, ROWS_, F_, D_);
    k_gemm<2, true, false><<<gD, blk, 0, stream>>>(
        mlp_h, W2_h + l * DF, b2 + (size_t)l * D_, h_f, h_f, nullptr, ROWS_, D_, F_);
  }

  // ---- head ----
  k_layernorm<<<B_, blk, 0, stream>>>(h_f, (size_t)S_ * D_, hlng, hlnb, cls_h);
  k_gemm<1, false, true><<<dim3(D_ / 128, 1), blk, 0, stream>>>(
      cls_h, hW1_h, hb1, nullptr, nullptr, c1_h, B_, D_, D_);
  k_gemm<0, true, false><<<dim3((NCLS_ + 127) / 128, 1), blk, 0, stream>>>(
      c1_h, hW2_h, hb2, nullptr, (float*)d_out, nullptr, B_, NCLS_, D_);
}
